// L2MultiheadAttention_42039139894036
// MI455X (gfx1250) — compile-verified, matrix operand order checked by reading, not run on hardware
//
#include <hip/hip_runtime.h>
#include <hip/hip_bf16.h>
#include <math.h>

// Problem constants (match reference)
#define T_DIM 1024
#define B_DIM 8
#define H_DIM 16
#define D_DIM 64
#define M_DIM 1024
#define G_DIM (T_DIM * B_DIM) // 8192 flattened (t,b) rows
#define INV_SQRT_D 0.125f     // 1/sqrt(64)

typedef __bf16 bf16_t;
typedef __attribute__((ext_vector_type(16))) __bf16 bf16x16;
typedef __attribute__((ext_vector_type(8)))  __bf16 bf16x8;
typedef __attribute__((ext_vector_type(8)))  float  floatx8;

// ---------------------------------------------------------------------------
// WMMA helpers (CDNA5 wave32 V_WMMA_F32_16X16X32_BF16)
// A fragment layout (16-bit, 16x32): lane<16 -> row=lane, K={k0..k0+7, k0+16..k0+23}
//                                    lane>=16 -> row=lane-16, K={k0+8..+15, k0+24..+31}
// B fragments use the same pattern on B^T rows (columns of B), so one loader
// serves row-major A and pre-transposed B.
// ---------------------------------------------------------------------------
static __device__ __forceinline__ bf16x16 frag_ld(const bf16_t* __restrict__ tile,
                                                  int ld, int k0) {
  const int lane = threadIdx.x & 31;
  const int r    = lane & 15;
  const int half = lane >> 4;
  const bf16_t* p0 = tile + (size_t)r * ld + k0 + half * 8;
  bf16x8 lo = *(const bf16x8*)(p0);       // 16B contiguous
  bf16x8 hi = *(const bf16x8*)(p0 + 16);  // 16B contiguous
  bf16x16 f;
#pragma unroll
  for (int i = 0; i < 8; ++i) { f[i] = lo[i]; f[i + 8] = hi[i]; }
  return f;
}

static __device__ __forceinline__ floatx8 wmma_bf16(bf16x16 a, bf16x16 b, floatx8 c) {
  // 8 args: (neg_a, A, neg_b, B, c_mod, C, reuse_a, reuse_b)
  return __builtin_amdgcn_wmma_f32_16x16x32_bf16(false, a, false, b, (short)0, c,
                                                 false, false);
}

// Register-blocked GEMM core: RT*16 rows x CT*16 cols per wave.
template <int RT, int CT>
static __device__ __forceinline__ void gemm_loop(const bf16_t* __restrict__ A, int lda,
                                                 const bf16_t* __restrict__ Bt, int ldb,
                                                 int K, floatx8 (&acc)[RT][CT]) {
#pragma unroll 2
  for (int k0 = 0; k0 < K; k0 += 32) {
    bf16x16 af[RT], bfr[CT];
#pragma unroll
    for (int r = 0; r < RT; ++r) af[r] = frag_ld(A + (size_t)(r * 16) * lda, lda, k0);
#pragma unroll
    for (int c = 0; c < CT; ++c) bfr[c] = frag_ld(Bt + (size_t)(c * 16) * ldb, ldb, k0);
#pragma unroll
    for (int r = 0; r < RT; ++r)
#pragma unroll
      for (int c = 0; c < CT; ++c)
        acc[r][c] = wmma_bf16(af[r], bfr[c], acc[r][c]);
  }
}

// ---------------------------------------------------------------------------
// Packing / small precompute kernels
// ---------------------------------------------------------------------------
__global__ void k_cvt_bf16(const float* __restrict__ src, bf16_t* __restrict__ dst, int n) {
  int i = blockIdx.x * blockDim.x + threadIdx.x;
  if (i < n) dst[i] = (bf16_t)src[i];
}

// q_weight [M,H,D] -> WqT [H][D][M] (B-ready for Q = X @ Wq[h])
__global__ void k_pack_wqT(const float* __restrict__ qw, bf16_t* __restrict__ wqT) {
  int i = blockIdx.x * blockDim.x + threadIdx.x;
  if (i >= M_DIM * H_DIM * D_DIM) return;
  int m = i / (H_DIM * D_DIM);
  int h = (i / D_DIM) % H_DIM;
  int d = i % D_DIM;
  wqT[(h * D_DIM + d) * M_DIM + m] = (bf16_t)qw[i];
}

// S1[h][e][d] = sum_m Wq[m][h][e] * Wv[m][h][d]   (fp32, tiny)
__global__ void k_s1(const float* __restrict__ qw, const float* __restrict__ vw,
                     float* __restrict__ s1) {
  int i = blockIdx.x * blockDim.x + threadIdx.x;
  if (i >= H_DIM * D_DIM * D_DIM) return;
  int h = i / (D_DIM * D_DIM);
  int e = (i / D_DIM) % D_DIM;
  int d = i % D_DIM;
  float acc = 0.f;
  for (int m = 0; m < M_DIM; ++m)
    acc += qw[(m * H_DIM + h) * D_DIM + e] * vw[(m * H_DIM + h) * D_DIM + d];
  s1[i] = acc;
}

// AVt[h][d][m] = (1/sqrt(D)) * sum_e Wq[m][h][e] * S1[h][e][d]  (B-ready for X @ AV[h])
__global__ void k_avt(const float* __restrict__ qw, const float* __restrict__ s1,
                      bf16_t* __restrict__ avT) {
  int i = blockIdx.x * blockDim.x + threadIdx.x;
  if (i >= H_DIM * D_DIM * M_DIM) return;
  int h = i / (D_DIM * M_DIM);
  int d = (i / M_DIM) % D_DIM;
  int m = i % M_DIM;
  float acc = 0.f;
#pragma unroll 8
  for (int e = 0; e < D_DIM; ++e)
    acc += qw[(m * H_DIM + h) * D_DIM + e] * s1[(h * D_DIM + e) * D_DIM + d];
  avT[i] = (bf16_t)(acc * INV_SQRT_D);
}

// ---------------------------------------------------------------------------
// Q = X @ Wq  -> Qbf [b][h][t][d] bf16.  One wave per 64x32 output tile.
// 128 row tiles x 32 col tiles = 4096 waves.
// ---------------------------------------------------------------------------
__global__ void __launch_bounds__(256) k_qproj(const bf16_t* __restrict__ Xbf,
                                               const bf16_t* __restrict__ wqT,
                                               bf16_t* __restrict__ Qbf) {
  const int wave = blockIdx.x * (blockDim.x >> 5) + (threadIdx.x >> 5);
  const int lane = threadIdx.x & 31;
  const int rt = wave >> 5; // 128 row tiles of 64 over G=8192
  const int ct = wave & 31; // 32 col tiles of 32 over H*D=1024
  const int h  = ct >> 1;
  const int d0 = (ct & 1) * 32;
  const bf16_t* Atile = Xbf + (size_t)(rt * 64) * M_DIM;
  const bf16_t* Btile = wqT + (size_t)(h * D_DIM + d0) * M_DIM;
  floatx8 acc[4][2] = {};
  gemm_loop<4, 2>(Atile, M_DIM, Btile, M_DIM, M_DIM, acc);
  const int n = lane & 15, half = lane >> 4;
#pragma unroll
  for (int r = 0; r < 4; ++r)
#pragma unroll
    for (int c = 0; c < 2; ++c) {
      const int d = d0 + c * 16 + n;
#pragma unroll
      for (int j = 0; j < 8; ++j) {
        int g = rt * 64 + r * 16 + half * 8 + j;
        int t = g >> 3, bb = g & 7;
        Qbf[((size_t)(bb * H_DIM + h) * T_DIM + t) * D_DIM + d] = (bf16_t)acc[r][c][j];
      }
    }
}

// q_sq[b][h][t] = sum_d Q^2
__global__ void k_qsq(const bf16_t* __restrict__ Qbf, float* __restrict__ qsq) {
  int i = blockIdx.x * blockDim.x + threadIdx.x;
  if (i >= B_DIM * H_DIM * T_DIM) return;
  const bf16_t* q = Qbf + (size_t)i * D_DIM;
  float acc = 0.f;
#pragma unroll
  for (int d = 0; d < D_DIM; ++d) { float v = (float)q[d]; acc += v * v; }
  qsq[i] = acc;
}

// XAV = X @ AV -> stored transposed XAVt [b][h][d][t]  (B-ready for P @ XAV)
__global__ void __launch_bounds__(256) k_xav(const bf16_t* __restrict__ Xbf,
                                             const bf16_t* __restrict__ avT,
                                             bf16_t* __restrict__ xavT) {
  const int wave = blockIdx.x * (blockDim.x >> 5) + (threadIdx.x >> 5);
  const int lane = threadIdx.x & 31;
  const int rt = wave >> 5;
  const int ct = wave & 31;
  const int h  = ct >> 1;
  const int d0 = (ct & 1) * 32;
  const bf16_t* Atile = Xbf + (size_t)(rt * 64) * M_DIM;
  const bf16_t* Btile = avT + (size_t)(h * D_DIM + d0) * M_DIM;
  floatx8 acc[4][2] = {};
  gemm_loop<4, 2>(Atile, M_DIM, Btile, M_DIM, M_DIM, acc);
  const int n = lane & 15, half = lane >> 4;
#pragma unroll
  for (int r = 0; r < 4; ++r)
#pragma unroll
    for (int c = 0; c < 2; ++c) {
      const int d = d0 + c * 16 + n;
#pragma unroll
      for (int j = 0; j < 8; ++j) {
        int g = rt * 64 + r * 16 + half * 8 + j;
        int t = g >> 3, bb = g & 7;
        xavT[((size_t)(bb * H_DIM + h) * D_DIM + d) * T_DIM + t] = (bf16_t)acc[r][c][j];
      }
    }
}

// ---------------------------------------------------------------------------
// Flash-style L2-distance attention + P@XAV.  One wave per (b,h, 32-row tile).
// logits = -(q_sq[t] + q_sq[s] - 2*Q·Qᵀ)/sqrt(D); online softmax over s.
// All 32 lanes own one query row each for the softmax statistics.
// Output Obf [g=t*8+b][h*64+d] bf16 (A-ready for final GEMM).
// ---------------------------------------------------------------------------
__global__ void __launch_bounds__(128) k_attn(const bf16_t* __restrict__ Qbf,
                                              const float* __restrict__ qsq,
                                              const bf16_t* __restrict__ xavT,
                                              bf16_t* __restrict__ Obf) {
  __shared__ float Lt[4][32][33];  // logits/probs tile per wave (32 t x 32 s)
  __shared__ float qrow[4][32];
  __shared__ float scl[4][32];
  __shared__ float lsum[4][32];
  const int w    = threadIdx.x >> 5;
  const int lane = threadIdx.x & 31;
  const int wave = blockIdx.x * 4 + w;
  const int rt = wave & 31;  // 32 row tiles of 32 over T
  const int bh = wave >> 5;  // b*H + h, 0..127
  const int t0 = rt * 32;
  const bf16_t* Q_bh  = Qbf  + (size_t)bh * T_DIM * D_DIM;
  const float*  qs_bh = qsq  + (size_t)bh * T_DIM;
  const bf16_t* XV_bh = xavT + (size_t)bh * D_DIM * T_DIM;
  const int n = lane & 15, half = lane >> 4;

  // Loop-invariant A fragments: two 16-row tiles x two K halves (D=64)
  bf16x16 aq[2][2];
#pragma unroll
  for (int r = 0; r < 2; ++r) {
    const bf16_t* Atile = Q_bh + (size_t)(t0 + r * 16) * D_DIM;
    aq[r][0] = frag_ld(Atile, D_DIM, 0);
    aq[r][1] = frag_ld(Atile, D_DIM, 32);
  }

  qrow[w][lane] = qs_bh[t0 + lane];
  __syncthreads();
  float qrow_m[2][8];
#pragma unroll
  for (int r = 0; r < 2; ++r)
#pragma unroll
    for (int j = 0; j < 8; ++j) qrow_m[r][j] = qrow[w][r * 16 + half * 8 + j];

  float m_i = -1e30f, l_i = 0.f;  // per-lane: stats of row t0+lane
  floatx8 o[2][4] = {};

  for (int s0 = 0; s0 < T_DIM; s0 += 32) {
    __syncthreads();  // prior iteration's Lt consumers are done

    // QK^T: two 16-wide s tiles x two 16-row q tiles (K = Q)
#pragma unroll
    for (int st = 0; st < 2; ++st) {
      const bf16_t* Bt = Q_bh + (size_t)(s0 + st * 16) * D_DIM;
      bf16x16 bk0 = frag_ld(Bt, D_DIM, 0);
      bf16x16 bk1 = frag_ld(Bt, D_DIM, 32);
      const float qcol = qs_bh[s0 + st * 16 + n];
#pragma unroll
      for (int r = 0; r < 2; ++r) {
        floatx8 cc = {};
        cc = wmma_bf16(aq[r][0], bk0, cc);
        cc = wmma_bf16(aq[r][1], bk1, cc);
#pragma unroll
        for (int j = 0; j < 8; ++j)
          Lt[w][r * 16 + half * 8 + j][st * 16 + n] =
              -(qrow_m[r][j] + qcol - 2.f * cc[j]) * INV_SQRT_D;
      }
    }
    __syncthreads();

    // Online softmax: one lane per query row, write p in place
    {
      float mb = -1e30f;
#pragma unroll
      for (int s = 0; s < 32; ++s) mb = fmaxf(mb, Lt[w][lane][s]);
      float m_new = fmaxf(m_i, mb);
      float sc = __expf(m_i - m_new);
      float ssum = 0.f;
#pragma unroll
      for (int s = 0; s < 32; ++s) {
        float p = __expf(Lt[w][lane][s] - m_new);
        Lt[w][lane][s] = p;
        ssum += p;
      }
      l_i = l_i * sc + ssum;
      m_i = m_new;
      scl[w][lane] = sc;
    }
    __syncthreads();

    // Rescale accumulators by per-row correction factor
#pragma unroll
    for (int r = 0; r < 2; ++r) {
      float sc_m[8];
#pragma unroll
      for (int j = 0; j < 8; ++j) sc_m[j] = scl[w][r * 16 + half * 8 + j];
#pragma unroll
      for (int dt = 0; dt < 4; ++dt)
#pragma unroll
        for (int j = 0; j < 8; ++j) o[r][dt][j] *= sc_m[j];
    }

    // Build P as A fragments straight from LDS
    bf16x16 pa[2];
#pragma unroll
    for (int r = 0; r < 2; ++r)
#pragma unroll
      for (int i = 0; i < 8; ++i) {
        pa[r][i]     = (bf16_t)Lt[w][r * 16 + n][half * 8 + i];
        pa[r][i + 8] = (bf16_t)Lt[w][r * 16 + n][16 + half * 8 + i];
      }

    // o[r][dt] += P(16x32) @ XAV(32x16); each B fragment reused by both r
#pragma unroll
    for (int dt = 0; dt < 4; ++dt) {
      bf16x16 bx = frag_ld(XV_bh + (size_t)(dt * 16) * T_DIM, T_DIM, s0);
#pragma unroll
      for (int r = 0; r < 2; ++r) o[r][dt] = wmma_bf16(pa[r], bx, o[r][dt]);
    }
  }

  lsum[w][lane] = l_i;
  __syncthreads();

  const int b = bh >> 4;
  const int h = bh & 15;
#pragma unroll
  for (int r = 0; r < 2; ++r) {
    float linv[8];
#pragma unroll
    for (int j = 0; j < 8; ++j) linv[j] = 1.f / lsum[w][r * 16 + half * 8 + j];
#pragma unroll
    for (int dt = 0; dt < 4; ++dt) {
      int d = dt * 16 + n;
#pragma unroll
      for (int j = 0; j < 8; ++j) {
        int t = t0 + r * 16 + half * 8 + j;
        Obf[((size_t)(t * B_DIM + b)) * (H_DIM * D_DIM) + h * D_DIM + d] =
            (bf16_t)(o[r][dt][j] * linv[j]);
      }
    }
  }
}

// ---------------------------------------------------------------------------
// out = O @ out_w^T + out_b   (out_w row-major [m][k] is already B-ready)
// One wave per 64x32 tile: 128 x 32 = 4096 waves.
// ---------------------------------------------------------------------------
__global__ void __launch_bounds__(256) k_out(const bf16_t* __restrict__ Obf,
                                             const bf16_t* __restrict__ Wbf,
                                             const float* __restrict__ bias,
                                             float* __restrict__ out) {
  const int wave = blockIdx.x * (blockDim.x >> 5) + (threadIdx.x >> 5);
  const int lane = threadIdx.x & 31;
  const int rt = wave >> 5;
  const int ct = wave & 31;
  const bf16_t* Atile = Obf + (size_t)(rt * 64) * M_DIM;
  const bf16_t* Btile = Wbf + (size_t)(ct * 32) * M_DIM;
  floatx8 acc[4][2] = {};
  gemm_loop<4, 2>(Atile, M_DIM, Btile, M_DIM, M_DIM, acc);
  const int n = lane & 15, half = lane >> 4;
#pragma unroll
  for (int c = 0; c < 2; ++c) {
    const int col = ct * 32 + c * 16 + n;
    const float bcol = bias[col];
#pragma unroll
    for (int r = 0; r < 4; ++r)
#pragma unroll
      for (int j = 0; j < 8; ++j) {
        int g = rt * 64 + r * 16 + half * 8 + j;
        out[(size_t)g * M_DIM + col] = acc[r][c][j] + bcol;
      }
  }
}

// ---------------------------------------------------------------------------
extern "C" void kernel_launch(void* const* d_in, const int* in_sizes, int n_in,
                              void* d_out, int out_size, void* d_ws, size_t ws_size,
                              hipStream_t stream) {
  (void)in_sizes; (void)n_in; (void)out_size; (void)ws_size;
  const float* x    = (const float*)d_in[0];
  const float* qw   = (const float*)d_in[1];
  const float* vw   = (const float*)d_in[2];
  const float* outw = (const float*)d_in[3];
  const float* outb = (const float*)d_in[4];
  float* out = (float*)d_out;

  char* ws = (char*)d_ws;
  size_t off = 0;
  auto alloc = [&](size_t bytes) -> void* {
    void* p = ws + off;
    off = (off + bytes + 255) & ~(size_t)255;
    return p;
  };
  bf16_t* Xbf  = (bf16_t*)alloc((size_t)G_DIM * M_DIM * 2);                 // 16 MB
  bf16_t* Wobf = (bf16_t*)alloc((size_t)M_DIM * M_DIM * 2);                 //  2 MB
  bf16_t* WqT  = (bf16_t*)alloc((size_t)H_DIM * D_DIM * M_DIM * 2);         //  2 MB
  float*  S1   = (float*) alloc((size_t)H_DIM * D_DIM * D_DIM * 4);         // 256 KB
  bf16_t* AVt  = (bf16_t*)alloc((size_t)H_DIM * D_DIM * M_DIM * 2);         //  2 MB
  bf16_t* Qbf  = (bf16_t*)alloc((size_t)B_DIM * H_DIM * T_DIM * D_DIM * 2); // 16 MB
  float*  Qsq  = (float*) alloc((size_t)B_DIM * H_DIM * T_DIM * 4);         // 512 KB
  bf16_t* XAVt = (bf16_t*)alloc((size_t)B_DIM * H_DIM * D_DIM * T_DIM * 2); // 16 MB
  bf16_t* Obf  = (bf16_t*)alloc((size_t)G_DIM * M_DIM * 2);                 // 16 MB

  const int nX = G_DIM * M_DIM;
  const int nW = M_DIM * M_DIM;
  k_cvt_bf16<<<(nX + 255) / 256, 256, 0, stream>>>(x, Xbf, nX);
  k_cvt_bf16<<<(nW + 255) / 256, 256, 0, stream>>>(outw, Wobf, nW);
  k_pack_wqT<<<(M_DIM * H_DIM * D_DIM + 255) / 256, 256, 0, stream>>>(qw, WqT);
  k_s1<<<(H_DIM * D_DIM * D_DIM + 255) / 256, 256, 0, stream>>>(qw, vw, S1);
  k_avt<<<(H_DIM * D_DIM * M_DIM + 255) / 256, 256, 0, stream>>>(qw, S1, AVt);

  // 128 row tiles * 32 col tiles = 4096 waves; 8 waves/block
  k_qproj<<<512, 256, 0, stream>>>(Xbf, WqT, Qbf);
  k_qsq<<<(B_DIM * H_DIM * T_DIM + 255) / 256, 256, 0, stream>>>(Qbf, Qsq);
  k_xav<<<512, 256, 0, stream>>>(Xbf, AVt, XAVt);

  // 128 (b,h) * 32 row tiles = 4096 waves; 4 waves/block
  k_attn<<<1024, 128, 0, stream>>>(Qbf, Qsq, XAVt, Obf);

  k_out<<<512, 256, 0, stream>>>(Obf, Wobf, outb, out);
}
